// BalancedTopKRouter_40982577938613
// MI455X (gfx1250) — matrix operand order repaired and checked by reading, not compiled, not run on hardware
//
#include <hip/hip_runtime.h>
#include <hip/hip_bf16.h>
#include <math.h>

// Problem constants (match reference setup_inputs)
#define T_TOKENS 16384
#define H_DIM    4096
#define N_EXP    64
#define TOPK_K   2
#define TOK_PER_BLK 128   // 8 waves * 16 tokens
#define THREADS  256

typedef float  v4f   __attribute__((ext_vector_type(4)));
typedef float  v8f   __attribute__((ext_vector_type(8)));
typedef __bf16 v8bf  __attribute__((ext_vector_type(8)));
typedef __bf16 v16bf __attribute__((ext_vector_type(16)));

// ---------------------------------------------------------------------------
// Kernel 1: convert gate weights f32 -> bf16 into workspace, zero accumulators
// acc layout (floats): [0..63]  = sum of router_probs per expert
//                      [64..127]= top-2 token counts per expert
//                      [128]    = sum of logsumexp^2
// ---------------------------------------------------------------------------
__global__ void moe_prep_kernel(const float* __restrict__ gw,
                                __bf16* __restrict__ wbf,
                                float* __restrict__ acc) {
    const int n = N_EXP * H_DIM;
    for (int idx = blockIdx.x * blockDim.x + threadIdx.x; idx < n;
         idx += gridDim.x * blockDim.x) {
        wbf[idx] = (__bf16)gw[idx];
    }
    if (blockIdx.x == 0 && threadIdx.x < 129) acc[threadIdx.x] = 0.0f;
}

// ---------------------------------------------------------------------------
// Kernel 2: WMMA gate projection + softmax + top-2 + loss partials
// ---------------------------------------------------------------------------
__global__ __launch_bounds__(THREADS) void
moe_router_kernel(const float* __restrict__ hs,
                  const __bf16* __restrict__ wbf,
                  float* __restrict__ out,
                  float* __restrict__ acc) {
    __shared__ float lds_logits[TOK_PER_BLK][N_EXP];   // 32 KB
    __shared__ float lds_cnt[N_EXP];
    __shared__ float lds_m[TOK_PER_BLK];
    __shared__ float lds_is[TOK_PER_BLK];
    __shared__ float lds_z;

    const int tid  = threadIdx.x;
    const int wave = tid >> 5;
    const int lane = tid & 31;
    const int hl   = lane >> 4;   // which half of the wave (0/1)
    const int l15  = lane & 15;

    // zero shared accumulators (visible after the sync below)
    if (tid < N_EXP) lds_cnt[tid] = 0.0f;
    if (tid == 0)    lds_z = 0.0f;

    const int tokenBase = blockIdx.x * TOK_PER_BLK + wave * 16;
    const float* aRow = hs + (size_t)(tokenBase + l15) * H_DIM;
    // B base for this lane: expert row (l15), offset by half-of-wave.
    const __bf16* bBase = wbf + (size_t)l15 * H_DIM + 8 * hl;

    v8f accf[4];
    #pragma unroll
    for (int nt = 0; nt < 4; ++nt)
        accf[nt] = (v8f){0.f, 0.f, 0.f, 0.f, 0.f, 0.f, 0.f, 0.f};

    // ---- software-pipelined A load (iteration k=0) ----
    // 16-bit A 16x32 layout: lane (hl,l15) element e: K = 16*(e>>3)+8*hl+(e&7)
    v4f a0 = *(const v4f*)(aRow + 8 * hl);
    v4f a1 = *(const v4f*)(aRow + 8 * hl + 4);
    v4f a2 = *(const v4f*)(aRow + 16 + 8 * hl);
    v4f a3 = *(const v4f*)(aRow + 16 + 8 * hl + 4);

    for (int k = 0; k < H_DIM; k += 32) {
        // ---- load ALL 4 B tiles into distinct fragments (one clause) ----
        // B 32x16 layout: lane holds column n = nt*16+l15, K = 16*(e>>3)+8*hl+(e&7),
        // i.e. 8 bf16 from W[n][k + 8*hl ..] and 8 from W[n][k + 16 + 8*hl ..]
        v16bf bfr[4];
        #pragma unroll
        for (int nt = 0; nt < 4; ++nt) {
            const __bf16* bPtr = bBase + (size_t)nt * 16 * H_DIM + k;
            v8bf b0 = *(const v8bf*)(bPtr);
            v8bf b1 = *(const v8bf*)(bPtr + 16);
            bfr[nt] = __builtin_shufflevector(
                b0, b1, 0, 1, 2, 3, 4, 5, 6, 7, 8, 9, 10, 11, 12, 13, 14, 15);
        }

        // ---- convert current A fragment to bf16 ----
        v16bf afrag;
        #pragma unroll
        for (int i = 0; i < 4; ++i) {
            afrag[i]      = (__bf16)a0[i];
            afrag[4 + i]  = (__bf16)a1[i];
            afrag[8 + i]  = (__bf16)a2[i];
            afrag[12 + i] = (__bf16)a3[i];
        }

        // ---- issue next iteration's A loads before the WMMA group ----
        if (k + 32 < H_DIM) {
            const float* aNext = aRow + k + 32;
            __builtin_prefetch(aNext + 256, 0, 3);   // global_prefetch_b8 ahead
            a0 = *(const v4f*)(aNext + 8 * hl);
            a1 = *(const v4f*)(aNext + 8 * hl + 4);
            a2 = *(const v4f*)(aNext + 16 + 8 * hl);
            a3 = *(const v4f*)(aNext + 16 + 8 * hl + 4);
        }

        // ---- 4 back-to-back WMMAs (independent accumulators/B regs) ----
        #pragma unroll
        for (int nt = 0; nt < 4; ++nt) {
            accf[nt] = __builtin_amdgcn_wmma_f32_16x16x32_bf16(
                false, afrag, false, bfr[nt], (short)0, accf[nt], false, false);
        }
    }

    // ---- scatter C to LDS: lane holds C[m = v + 8*hl][n = l15] per tile ----
    #pragma unroll
    for (int nt = 0; nt < 4; ++nt) {
        #pragma unroll
        for (int v = 0; v < 8; ++v) {
            lds_logits[wave * 16 + v + 8 * hl][nt * 16 + l15] = accf[nt][v];
        }
    }
    __syncthreads();

    // ---- phase 2: per-token softmax, top-2, renorm, loss partials ----
    if (tid < TOK_PER_BLK) {
        const float* row = lds_logits[tid];
        float m = -INFINITY;
        #pragma unroll 8
        for (int e = 0; e < N_EXP; ++e) m = fmaxf(m, row[e]);

        float s = 0.0f;
        float v1 = -INFINITY, v2 = -INFINITY;
        int e1 = 0, e2 = 0;
        for (int e = 0; e < N_EXP; ++e) {
            float l = row[e];
            s += __expf(l - m);
            if (l > v1)      { v2 = v1; e2 = e1; v1 = l; e1 = e; }
            else if (l > v2) { v2 = l; e2 = e; }
        }
        float invs = 1.0f / s;
        float logz = m + __logf(s);
        lds_m[tid]  = m;
        lds_is[tid] = invs;

        float p1 = __expf(v1 - m) * invs;
        float p2 = __expf(v2 - m) * invs;
        float denom = p1 + p2 + 1e-9f;

        int t = blockIdx.x * TOK_PER_BLK + tid;
        out[2 * t]     = p1 / denom;                 // expert_weights
        out[2 * t + 1] = p2 / denom;
        out[T_TOKENS * 2 + 2 * t]     = (float)e1;   // expert_indices (as f32)
        out[T_TOKENS * 2 + 2 * t + 1] = (float)e2;

        atomicAdd(&lds_cnt[e1], 1.0f);
        atomicAdd(&lds_cnt[e2], 1.0f);
        atomicAdd(&lds_z, logz * logz);
    }
    __syncthreads();

    // per-expert prob sums (column reduction over this block's tokens)
    if (tid < N_EXP) {
        float psum = 0.0f;
        for (int t = 0; t < TOK_PER_BLK; ++t)
            psum += __expf(lds_logits[t][tid] - lds_m[t]) * lds_is[t];
        atomicAdd(&acc[tid], psum);
        atomicAdd(&acc[64 + tid], lds_cnt[tid]);
    }
    if (tid == 0) atomicAdd(&acc[128], lds_z);
}

// ---------------------------------------------------------------------------
// Kernel 3: finalize losses + expert utilization
// ---------------------------------------------------------------------------
__global__ void moe_final_kernel(const float* __restrict__ acc,
                                 float* __restrict__ out) {
    __shared__ float red[N_EXP];
    int e = threadIdx.x;
    float cnt = acc[64 + e];
    float f = cnt / (float)(T_TOKENS * TOPK_K);
    float P = acc[e] / (float)T_TOKENS;
    red[e] = f * P;
    out[T_TOKENS * 4 + 2 + e] = f;   // expert_util
    __syncthreads();
    if (e == 0) {
        float s = 0.0f;
        for (int i = 0; i < N_EXP; ++i) s += red[i];
        out[T_TOKENS * 4]     = 0.01f * (float)N_EXP * s;              // aux loss
        out[T_TOKENS * 4 + 1] = 0.001f * (acc[128] / (float)T_TOKENS); // z loss
    }
}

// ---------------------------------------------------------------------------
extern "C" void kernel_launch(void* const* d_in, const int* in_sizes, int n_in,
                              void* d_out, int out_size, void* d_ws, size_t ws_size,
                              hipStream_t stream) {
    const float* hidden = (const float*)d_in[0];   // [T, H] f32
    const float* gatew  = (const float*)d_in[1];   // [E, H] f32
    float* out = (float*)d_out;

    // workspace: bf16 weights (512 KB) then 129-float accumulator block
    __bf16* wbf = (__bf16*)d_ws;
    float*  acc = (float*)((char*)d_ws + (size_t)N_EXP * H_DIM * sizeof(__bf16));

    moe_prep_kernel<<<256, THREADS, 0, stream>>>(gatew, wbf, acc);
    moe_router_kernel<<<T_TOKENS / TOK_PER_BLK, THREADS, 0, stream>>>(hidden, wbf, out, acc);
    moe_final_kernel<<<1, N_EXP, 0, stream>>>(acc, out);
}
